// GMM_74560632258698
// MI455X (gfx1250) — matrix operand order repaired and checked by reading, not compiled, not run on hardware
//
#include <hip/hip_runtime.h>
#include <math.h>

typedef __attribute__((ext_vector_type(16))) _Float16 v16h;
typedef __attribute__((ext_vector_type(8)))  _Float16 v8h;
typedef __attribute__((ext_vector_type(8)))  float    v8f;
typedef __attribute__((ext_vector_type(4)))  float    v4f;

#define NROWS 262144
#define KC    256
#define DD    128
#define LOG2PI 1.8378770664093453f
#define SCALEF 0.1f

// ---------------------------------------------------------------------------
// Phase 1: build f16 coefficient matrices and per-component constants.
//   W1[k,d] = exp(-2*log_std)            (inverse variance)
//   W2[k,d] = -2 * mu * ivar
//   constk[k] = -0.5*(sum mu^2*ivar + D*log2pi) - sum log_std + weights[k]
// (logsumexp(weights) is subtracted by prep_lse.)
// ---------------------------------------------------------------------------
__global__ void prep_w(const float* __restrict__ means,
                       const float* __restrict__ log_stds,
                       const float* __restrict__ weights,
                       _Float16* __restrict__ W1, _Float16* __restrict__ W2,
                       float* __restrict__ constk) {
    const int k = blockIdx.x;
    const int d = threadIdx.x;                  // 0..127
    const size_t off = (size_t)k * DD + d;
    const float mu = means[off];
    const float ls = log_stds[off];
    const float iv = __expf(-2.0f * ls);
    W1[off] = (_Float16)iv;
    W2[off] = (_Float16)(-2.0f * mu * iv);

    __shared__ float sA[DD];
    __shared__ float sB[DD];
    sA[d] = mu * mu * iv;
    sB[d] = ls;
    __syncthreads();
    for (int s = DD / 2; s > 0; s >>= 1) {
        if (d < s) { sA[d] += sA[d + s]; sB[d] += sB[d + s]; }
        __syncthreads();
    }
    if (d == 0)
        constk[k] = -0.5f * (sA[0] + (float)DD * LOG2PI) - sB[0] + weights[k];
}

__global__ void prep_lse(const float* __restrict__ weights,
                         float* __restrict__ constk,
                         float* __restrict__ qslot) {
    const int k = threadIdx.x;                  // 0..255
    __shared__ float sm[KC];
    const float w = weights[k];
    sm[k] = w;
    __syncthreads();
    for (int s = KC / 2; s > 0; s >>= 1) {
        if (k < s) sm[k] = fmaxf(sm[k], sm[k + s]);
        __syncthreads();
    }
    const float mx = sm[0];
    __syncthreads();
    sm[k] = __expf(w - mx);
    __syncthreads();
    for (int s = KC / 2; s > 0; s >>= 1) {
        if (k < s) sm[k] += sm[k + s];
        __syncthreads();
    }
    const float lse = mx + __logf(sm[0]);
    constk[k] -= lse;
    if (k == 0) *qslot = 0.0f;
}

// ---------------------------------------------------------------------------
// Main fused kernel. Block = 256 threads = 8 waves; each wave owns 16 rows
// and all 256 components (16 WMMA accumulator tiles). Reduction depth 256
// is realized as 4 d-chunks x {x^2 vs W1, x vs W2} WMMAs.
// Streaming traffic (x, noise, outputs) uses non-temporal cache policy so
// the 128 KB W1/W2 + means/log_stds working set stays resident in WGP$/L2.
// ---------------------------------------------------------------------------
__launch_bounds__(256)
__global__ void gmm_main(const float* __restrict__ x,
                         const float* __restrict__ noise,
                         const float* __restrict__ means,
                         const float* __restrict__ log_stds,
                         const _Float16* __restrict__ W1,
                         const _Float16* __restrict__ W2,
                         const float* __restrict__ constk,
                         float* __restrict__ out_x,
                         float* __restrict__ out_idx,
                         float* __restrict__ out_q) {
    const int lane = threadIdx.x & 31;
    const int wave = threadIdx.x >> 5;
    const int l15  = lane & 15;
    const int g    = lane >> 4;                 // half-wave group
    const int m0   = blockIdx.x * 128 + wave * 16;

    v8f acc[16];
    const v8f vzero = {0.f, 0.f, 0.f, 0.f, 0.f, 0.f, 0.f, 0.f};
#pragma unroll
    for (int t = 0; t < 16; ++t) acc[t] = vzero;

    // A operand: row = m0 + l15 (contiguous k order used here: per-lane half covers
    // tile cols g*16..g*16+15, matching B).
    const float* xrow = x + (size_t)(m0 + l15) * DD;

#pragma unroll
    for (int dc = 0; dc < 4; ++dc) {
        const int c0 = dc * 32 + g * 16;
        const v4f f0 = __builtin_nontemporal_load((const v4f*)(xrow + c0));
        const v4f f1 = __builtin_nontemporal_load((const v4f*)(xrow + c0 + 4));
        const v4f f2 = __builtin_nontemporal_load((const v4f*)(xrow + c0 + 8));
        const v4f f3 = __builtin_nontemporal_load((const v4f*)(xrow + c0 + 12));

        v16h a_lin, a_sq;
#pragma unroll
        for (int i = 0; i < 4; ++i) {
            const float x0 = f0[i], x1 = f1[i], x2 = f2[i], x3 = f3[i];
            a_lin[i]      = (_Float16)x0;
            a_lin[4 + i]  = (_Float16)x1;
            a_lin[8 + i]  = (_Float16)x2;
            a_lin[12 + i] = (_Float16)x3;
            a_sq[i]       = (_Float16)(x0 * x0);
            a_sq[4 + i]   = (_Float16)(x1 * x1);
            a_sq[8 + i]   = (_Float16)(x2 * x2);
            a_sq[12 + i]  = (_Float16)(x3 * x3);
        }

        // B operand: lane = component column (l15), per-lane halfs run along
        // 16 consecutive reduction values: d = dc*32 + g*16 + 0..15.
        // Regular (temporal) loads: W1/W2 are hot in WGP$/L2.
        const int bcol = dc * 32 + g * 16;
#pragma unroll
        for (int t = 0; t < 16; ++t) {
            const size_t wroff = (size_t)(t * 16 + l15) * DD + bcol;
            const v16h b1 = *(const v16h*)(W1 + wroff);
            acc[t] = __builtin_amdgcn_wmma_f32_16x16x32_f16(
                false, a_sq, false, b1, (short)0, acc[t], false, false);
            const v16h b2 = *(const v16h*)(W2 + wroff);
            acc[t] = __builtin_amdgcn_wmma_f32_16x16x32_f16(
                false, a_lin, false, b2, (short)0, acc[t], false, false);
        }
    }

    // Per-lane component constants: k = t*16 + l15.
    float ck[16];
#pragma unroll
    for (int t = 0; t < 16; ++t) ck[t] = constk[t * 16 + l15];

    // Epilogue: D-tile element (M = r + 8*g, N = l15) -> row m0+8g+r, comp t*16+l15.
    float qsum = 0.0f;
#pragma unroll
    for (int r = 0; r < 8; ++r) {
        const int m = m0 + 8 * g + r;

        // max / argmax over k (local over t, then 16-lane butterfly)
        float mx = -3.4e38f;
        int   mi = 0;
#pragma unroll
        for (int t = 0; t < 16; ++t) {
            const float j = -0.5f * acc[t][r] + ck[t];
            if (j > mx) { mx = j; mi = t * 16 + l15; }
        }
#pragma unroll
        for (int s = 1; s < 16; s <<= 1) {
            const float omx = __shfl_xor(mx, s, 32);
            const int   omi = __shfl_xor(mi, s, 32);
            if (omx > mx || (omx == mx && omi < mi)) { mx = omx; mi = omi; }
        }

        // softmax sums: s1 = sum exp(j-mx), s2 = sum exp(j-mx)*j
        float s1 = 0.0f, s2 = 0.0f;
#pragma unroll
        for (int t = 0; t < 16; ++t) {
            const float j = -0.5f * acc[t][r] + ck[t];
            const float e = __expf(j - mx);
            s1 += e;
            s2 += e * j;
        }
#pragma unroll
        for (int s = 1; s < 16; s <<= 1) {
            s1 += __shfl_xor(s1, s, 32);
            s2 += __shfl_xor(s2, s, 32);
        }

        if (l15 == 0) {
            qsum += s2 / s1;
            __builtin_nontemporal_store((float)mi, out_idx + m);
        }

        // Resample row m from component mi: 16 lanes x 8 elements each.
        // means/log_stds gathers are temporal (hot, 256 KB); noise/out are
        // streaming non-temporal.
        {
            const float* mrow = means    + (size_t)mi * DD;
            const float* srow = log_stds + (size_t)mi * DD;
            const float* nrow = noise    + (size_t)m  * DD;
            float*       orow = out_x    + (size_t)m  * DD;
            const int dbase = l15 * 8;
#pragma unroll
            for (int i = 0; i < 2; ++i) {
                const v4f nv = __builtin_nontemporal_load(
                    (const v4f*)(nrow + dbase + 4 * i));
                v4f ov;
#pragma unroll
                for (int j2 = 0; j2 < 4; ++j2) {
                    const int d = dbase + 4 * i + j2;
                    ov[j2] = mrow[d] + __expf(srow[d]) * nv[j2] * SCALEF;
                }
                __builtin_nontemporal_store(ov, (v4f*)(orow + dbase + 4 * i));
            }
        }
    }

    // Q: combine the two half-wave partial sums, one atomic per wave.
    const float other = __shfl_xor(qsum, 16, 32);
    if (lane == 0)
        atomicAdd(out_q, (qsum + other) * (1.0f / ((float)NROWS * (float)KC)));
}

// ---------------------------------------------------------------------------
extern "C" void kernel_launch(void* const* d_in, const int* in_sizes, int n_in,
                              void* d_out, int out_size, void* d_ws, size_t ws_size,
                              hipStream_t stream) {
    const float* x        = (const float*)d_in[0];   // [N, D]
    const float* means    = (const float*)d_in[1];   // [K, D]
    const float* log_stds = (const float*)d_in[2];   // [K, D]
    const float* weights  = (const float*)d_in[3];   // [K]
    const float* noise    = (const float*)d_in[4];   // [N, D]

    float* out     = (float*)d_out;
    float* out_x   = out;                                // N*D
    float* out_idx = out + (size_t)NROWS * DD;           // N
    float* out_q   = out_idx + NROWS;                    // 1

    _Float16* W1     = (_Float16*)d_ws;                  // K*D halfs (64 KB)
    _Float16* W2     = W1 + (size_t)KC * DD;             // K*D halfs (64 KB)
    float*    constk = (float*)(W2 + (size_t)KC * DD);   // K floats

    prep_w  <<<KC, DD, 0, stream>>>(means, log_stds, weights, W1, W2, constk);
    prep_lse<<<1,  KC, 0, stream>>>(weights, constk, out_q);
    gmm_main<<<NROWS / 128, 256, 0, stream>>>(x, noise, means, log_stds,
                                              W1, W2, constk,
                                              out_x, out_idx, out_q);
}
